// GLANTConv_38998303048289
// MI455X (gfx1250) — matrix operand order repaired and checked by reading, not compiled, not run on hardware
//
#include <hip/hip_runtime.h>
#include <hip/hip_bf16.h>

typedef __attribute__((ext_vector_type(16))) _Float16 v16h;
typedef __attribute__((ext_vector_type(8)))  _Float16 v8h;
typedef __attribute__((ext_vector_type(8)))  float    v8f;

#define IN_CH     128
#define HC        256   // HEADS * OUT_CH
#define OUTC      64
#define HEADS     4
#define NEG_SLOPE 0.2f

// ---------- order-preserving float <-> uint map (for atomic max on floats) ----------
__device__ __forceinline__ unsigned f2ord(float f) {
  unsigned b = __float_as_uint(f);
  return (b & 0x80000000u) ? ~b : (b | 0x80000000u);
}
__device__ __forceinline__ float ord2f(unsigned u) {
  return (u & 0x80000000u) ? __uint_as_float(u ^ 0x80000000u) : __uint_as_float(~u);
}
#define ORD_NEG_INF 0x007FFFFFu   // f2ord(-inf)

// ---------- conversion kernels ----------
__global__ void k_cvt_x_f16(const float* __restrict__ x, _Float16* __restrict__ xh, int n) {
  int i = blockIdx.x * blockDim.x + threadIdx.x;
  if (i < n) xh[i] = (_Float16)x[i];
}

// Wt[n*128 + k] = W[k*256 + n]  (transpose so K is contiguous per output column)
__global__ void k_cvt_w_t(const float* __restrict__ W, _Float16* __restrict__ Wt) {
  int i = blockIdx.x * blockDim.x + threadIdx.x;   // i = n*128 + k
  if (i < IN_CH * HC) {
    int n = i >> 7, k = i & 127;
    Wt[i] = (_Float16)W[k * HC + n];
  }
}

// ---------- WMMA GEMM: C[M,256](f32) = A[M,128](f16) * W[128,256], W transposed ----------
// One wave computes a 16(row) x 64(col) strip: 4 column tiles share each A fragment,
// so every A load feeds 4 v_wmma instructions (4 K-steps x 4 tiles = 16 WMMA / wave).
__global__ __launch_bounds__(256) void k_gemm_f16_wmma(
    const _Float16* __restrict__ A, const _Float16* __restrict__ Bt,
    float* __restrict__ C, int M) {
  const int wid  = blockIdx.x * 8 + (threadIdx.x >> 5);
  const int lane = threadIdx.x & 31;
  const int row_tiles = (M + 15) >> 4;
  const int mt = wid >> 2;          // row tile
  const int ng = wid & 3;           // column group: columns [ng*64, ng*64+64)
  if (mt >= row_tiles) return;      // wave-uniform: EXEC all-ones at WMMA
  const int half16 = lane >> 4;     // which K-subset this lane carries
  const int l15    = lane & 15;
  const _Float16* arow = A + (size_t)(mt * 16 + l15) * IN_CH;       // A row M = lane&15
  const _Float16* b0 = Bt + (size_t)(ng * 64 +  0 + l15) * IN_CH;   // B col N = lane&15
  const _Float16* b1 = Bt + (size_t)(ng * 64 + 16 + l15) * IN_CH;
  const _Float16* b2 = Bt + (size_t)(ng * 64 + 32 + l15) * IN_CH;
  const _Float16* b3 = Bt + (size_t)(ng * 64 + 48 + l15) * IN_CH;
  v8f acc0 = {}, acc1 = {}, acc2 = {}, acc3 = {};
#pragma unroll
  for (int k0 = 0; k0 < IN_CH; k0 += 32) {
    union { v16h v; v8h h[2]; } a;
    // A 16x32 f16 layout: elems 0..15 -> K = k0 + half16*16 + i (contiguous, matches B)
    a.h[0] = *(const v8h*)(arow + k0 + half16 * 16);
    a.h[1] = *(const v8h*)(arow + k0 + half16 * 16 + 8);
    // B 32x16 f16 layout: elems 0..15 -> K = k0 + half16*16 + i (contiguous, transposed W)
    const int bo = k0 + half16 * 16;
    v16h bv0 = *(const v16h*)(b0 + bo);
    v16h bv1 = *(const v16h*)(b1 + bo);
    v16h bv2 = *(const v16h*)(b2 + bo);
    v16h bv3 = *(const v16h*)(b3 + bo);
    acc0 = __builtin_amdgcn_wmma_f32_16x16x32_f16(false, a.v, false, bv0, (short)0, acc0, false, false);
    acc1 = __builtin_amdgcn_wmma_f32_16x16x32_f16(false, a.v, false, bv1, (short)0, acc1, false, false);
    acc2 = __builtin_amdgcn_wmma_f32_16x16x32_f16(false, a.v, false, bv2, (short)0, acc2, false, false);
    acc3 = __builtin_amdgcn_wmma_f32_16x16x32_f16(false, a.v, false, bv3, (short)0, acc3, false, false);
  }
  // D layout: VGPR j -> M = half16*8 + j, N = lane&15
  float* crow = C + (size_t)(mt * 16 + half16 * 8) * HC + ng * 64 + l15;
#pragma unroll
  for (int j = 0; j < 8; ++j) {
    crow[(size_t)j * HC +  0] = acc0[j];
    crow[(size_t)j * HC + 16] = acc1[j];
    crow[(size_t)j * HC + 32] = acc2[j];
    crow[(size_t)j * HC + 48] = acc3[j];
  }
}

// ---------- per-hop init ----------
__global__ void k_init(float* __restrict__ agg, float* __restrict__ denom,
                       unsigned* __restrict__ emax, int n_nodes) {
  int i = blockIdx.x * blockDim.x + threadIdx.x;
  if (i < n_nodes * HC) agg[i] = 0.f;
  if (i < n_nodes * HEADS) { denom[i] = 0.f; emax[i] = ORD_NEG_INF; }
}

// ---------- pass 1: per-edge attention logits + segment max ----------
// one wave32 per edge; 8 lanes per head; each lane covers 8 contiguous channels
__global__ __launch_bounds__(256) void k_edge_scores(
    const int* __restrict__ src, const int* __restrict__ dst,
    int E_real, int n_nodes, int with_self,
    const float* __restrict__ xl, const float* __restrict__ xr,
    const float* __restrict__ att,
    float* __restrict__ e, unsigned* __restrict__ emax) {
  int wid  = blockIdx.x * 8 + (threadIdx.x >> 5);
  int lane = threadIdx.x & 31;
  int total = E_real + (with_self ? n_nodes : 0);
  if (wid >= total) return;
  int s, d;
  if (wid < E_real) { s = src[wid]; d = dst[wid]; } else { s = d = wid - E_real; }
  const float4* xs = (const float4*)(xl + (size_t)s * HC) + lane * 2;
  const float4* xd = (const float4*)(xr + (size_t)d * HC) + lane * 2;
  const float4* at = (const float4*)att + lane * 2;
  float sum = 0.f;
#pragma unroll
  for (int q = 0; q < 2; ++q) {
    float4 a = xs[q], b = xd[q], w = at[q];
    float mx = a.x + b.x, my = a.y + b.y, mz = a.z + b.z, mw = a.w + b.w;
    mx = mx > 0.f ? mx : NEG_SLOPE * mx;
    my = my > 0.f ? my : NEG_SLOPE * my;
    mz = mz > 0.f ? mz : NEG_SLOPE * mz;
    mw = mw > 0.f ? mw : NEG_SLOPE * mw;
    sum = fmaf(mx, w.x, fmaf(my, w.y, fmaf(mz, w.z, fmaf(mw, w.w, sum))));
  }
  // reduce across the 8 lanes of this head
  sum += __shfl_xor(sum, 1, 32);
  sum += __shfl_xor(sum, 2, 32);
  sum += __shfl_xor(sum, 4, 32);
  int head = lane >> 3;
  if ((lane & 7) == 0) {
    e[(size_t)wid * HEADS + head] = sum;
    atomicMax(&emax[(size_t)d * HEADS + head], f2ord(sum));
  }
}

// ---------- pass 2: exp, denom, weighted scatter-add of xl[src] ----------
__global__ __launch_bounds__(256) void k_edge_aggregate(
    const int* __restrict__ src, const int* __restrict__ dst,
    int E_real, int n_nodes, int with_self,
    const float* __restrict__ xl, const float* __restrict__ e,
    const unsigned* __restrict__ emax,
    float* __restrict__ denom, float* __restrict__ agg) {
  int wid  = blockIdx.x * 8 + (threadIdx.x >> 5);
  int lane = threadIdx.x & 31;
  int total = E_real + (with_self ? n_nodes : 0);
  if (wid >= total) return;
  int s, d;
  if (wid < E_real) { s = src[wid]; d = dst[wid]; } else { s = d = wid - E_real; }
  int head = lane >> 3;
  float ev = e[(size_t)wid * HEADS + head];
  float mx = ord2f(emax[(size_t)d * HEADS + head]);
  float ex = __expf(ev - mx);
  const float4* xs = (const float4*)(xl + (size_t)s * HC) + lane * 2;
  float* ag = agg + (size_t)d * HC + lane * 8;
  float4 a0 = xs[0], a1 = xs[1];
  atomicAdd(ag + 0, ex * a0.x);
  atomicAdd(ag + 1, ex * a0.y);
  atomicAdd(ag + 2, ex * a0.z);
  atomicAdd(ag + 3, ex * a0.w);
  atomicAdd(ag + 4, ex * a1.x);
  atomicAdd(ag + 5, ex * a1.y);
  atomicAdd(ag + 6, ex * a1.z);
  atomicAdd(ag + 7, ex * a1.w);
  if ((lane & 7) == 0) atomicAdd(&denom[(size_t)d * HEADS + head], ex);
}

// ---------- finalize: normalize, mean over heads, write / gated-add ----------
__global__ void k_finalize(const float* __restrict__ agg, const float* __restrict__ denom,
                           float* __restrict__ out, int n_nodes, int add_mode,
                           const float* __restrict__ theta) {
  int idx = blockIdx.x * blockDim.x + threadIdx.x;     // n*64 + c
  if (idx >= n_nodes * OUTC) return;
  int n = idx >> 6, c = idx & 63;
  float acc = 0.f;
#pragma unroll
  for (int h = 0; h < HEADS; ++h) {
    float dn = denom[n * HEADS + h];
    float v  = agg[(size_t)n * HC + h * OUTC + c];
    acc += (dn > 0.f) ? v / dn : 0.f;
  }
  acc *= 0.25f;   // mean over heads
  if (add_mode) {
    float t  = theta[0];
    float sc = 1.f / (1.f + __expf(-t));
    out[idx] += sc * acc;
  } else {
    out[idx] = acc;
  }
}

extern "C" void kernel_launch(void* const* d_in, const int* in_sizes, int n_in,
                              void* d_out, int out_size, void* d_ws, size_t ws_size,
                              hipStream_t stream) {
  const float* x     = (const float*)d_in[0];
  const int*   ei1   = (const int*)d_in[1];
  const int*   ei2   = (const int*)d_in[2];
  const float* W_l   = (const float*)d_in[3];
  const float* W_r0  = (const float*)d_in[4];
  const float* W_r1  = (const float*)d_in[5];
  const float* att0  = (const float*)d_in[6];
  const float* att1  = (const float*)d_in[7];
  const float* theta = (const float*)d_in[8];
  float* out = (float*)d_out;

  const int N  = in_sizes[0] / IN_CH;
  const int E1 = in_sizes[1] / 2;
  const int E2 = in_sizes[2] / 2;
  const int Emax = (E1 > E2 ? E1 : E2);

  // carve workspace
  char* p = (char*)d_ws;
  auto carve = [&](size_t bytes) { char* r = p; p += (bytes + 255) & ~(size_t)255; return (void*)r; };
  _Float16* xh   = (_Float16*)carve((size_t)N * IN_CH * 2);
  _Float16* wtl  = (_Float16*)carve((size_t)IN_CH * HC * 2);
  _Float16* wtr0 = (_Float16*)carve((size_t)IN_CH * HC * 2);
  _Float16* wtr1 = (_Float16*)carve((size_t)IN_CH * HC * 2);
  float*    xl   = (float*)carve((size_t)N * HC * 4);
  float*    xr   = (float*)carve((size_t)N * HC * 4);       // reused per hop
  float*    agg  = (float*)carve((size_t)N * HC * 4);
  float*    ebuf = (float*)carve((size_t)(Emax + N) * HEADS * 4);
  unsigned* emax = (unsigned*)carve((size_t)N * HEADS * 4);
  float*    den  = (float*)carve((size_t)N * HEADS * 4);

  const int TB = 256;
  // conversions
  k_cvt_x_f16<<<(N * IN_CH + TB - 1) / TB, TB, 0, stream>>>(x, xh, N * IN_CH);
  k_cvt_w_t<<<(IN_CH * HC + TB - 1) / TB, TB, 0, stream>>>(W_l,  wtl);
  k_cvt_w_t<<<(IN_CH * HC + TB - 1) / TB, TB, 0, stream>>>(W_r0, wtr0);
  k_cvt_w_t<<<(IN_CH * HC + TB - 1) / TB, TB, 0, stream>>>(W_r1, wtr1);

  const int row_tiles  = (N + 15) / 16;
  const int gemm_waves = row_tiles * 4;                  // 4 column groups of 64
  const int gemm_blks  = (gemm_waves + 7) / 8;

  // xl = xh @ W_l (shared across hops)
  k_gemm_f16_wmma<<<gemm_blks, TB, 0, stream>>>(xh, wtl, xl, N);

  const int init_blks = (N * HC + TB - 1) / TB;
  const int fin_blks  = (N * OUTC + TB - 1) / TB;

  // ---- hop 0: edges1 + self loops, att0, W_r0 ----
  {
    k_gemm_f16_wmma<<<gemm_blks, TB, 0, stream>>>(xh, wtr0, xr, N);
    k_init<<<init_blks, TB, 0, stream>>>(agg, den, emax, N);
    int total = E1 + N;
    int eblks = (total + 7) / 8;
    k_edge_scores   <<<eblks, TB, 0, stream>>>(ei1, ei1 + E1, E1, N, 1, xl, xr, att0, ebuf, emax);
    k_edge_aggregate<<<eblks, TB, 0, stream>>>(ei1, ei1 + E1, E1, N, 1, xl, ebuf, emax, den, agg);
    k_finalize<<<fin_blks, TB, 0, stream>>>(agg, den, out, N, /*add=*/0, theta);
  }

  // ---- hop 1: edges2 (no self loops), att1, W_r1, gated residual ----
  {
    k_gemm_f16_wmma<<<gemm_blks, TB, 0, stream>>>(xh, wtr1, xr, N);
    k_init<<<init_blks, TB, 0, stream>>>(agg, den, emax, N);
    int total = E2;
    int eblks = (total + 7) / 8;
    k_edge_scores   <<<eblks, TB, 0, stream>>>(ei2, ei2 + E2, E2, N, 0, xl, xr, att1, ebuf, emax);
    k_edge_aggregate<<<eblks, TB, 0, stream>>>(ei2, ei2 + E2, E2, N, 0, xl, ebuf, emax, den, agg);
    k_finalize<<<fin_blks, TB, 0, stream>>>(agg, den, out, N, /*add=*/1, theta);
  }
}
